// Actor_67791763800611
// MI455X (gfx1250) — matrix operand order repaired and checked by reading, not compiled, not run on hardware
//
#include <hip/hip_runtime.h>
#include <hip/hip_bf16.h>
#include <math.h>

typedef __attribute__((ext_vector_type(16))) _Float16 v16h;
typedef __attribute__((ext_vector_type(8)))  _Float16 v8h;
typedef __attribute__((ext_vector_type(8)))  float    v8f;

#define NNODES 21
#define PI_F 3.14159265358979323846f

// ---- WMMA fragment helpers (gfx1250 wave32 layouts, 16x16x32 f16) ----------
// A: 16x32 (MxK), row-major f16 source with 16B-aligned rows.
// lane L: m = L&15, g = L>>4; halves[2v+t] = K 16g+2v+t
//  -> two contiguous 8-half (16B) chunks at K = kb+16g and K = kb+16g+8.
__device__ __forceinline__ v16h load_a_lds(const _Float16* base, int m,
                                           int stride, int kb, int g) {
  const _Float16* p = base + m * stride + kb + 16 * g;
  v8h lo = *(const v8h*)p;           // ds_load_b128
  v8h hi = *(const v8h*)(p + 8);     // ds_load_b128
  return __builtin_shufflevector(lo, hi, 0, 1, 2, 3, 4, 5, 6, 7,
                                 8, 9, 10, 11, 12, 13, 14, 15);
}

// B: 32x16 (KxN) from PACKED TRANSPOSED f16 weights WT[N][K] (K contiguous).
// lane L: n = L&15, g = L>>4; halves[2v+t] = K 16g+2v+t
//  -> one contiguous 16-half (32B) chunk at WT[n][kb+16g].
__device__ __forceinline__ v16h load_b_t16(const _Float16* WT, int ldk,
                                           int kb, int n, int g) {
  return *(const v16h*)(WT + n * ldk + kb + 16 * g);  // 2x global_load_b128
}

__device__ __forceinline__ v8f wmma_f16(v16h a, v16h b, v8f c) {
  return __builtin_amdgcn_wmma_f32_16x16x32_f16(false, a, false, b,
                                                (short)0, c, false, false);
}

// ============================================================================
// Kernel 0: pack all GEMM weights to f16, transposed [N][K] (K-contiguous).
//   Wl2,Wr2: [256,64] -> [64][256];  fc1: [64,256] -> [256][64];
//   fc2: [256,256] -> [256][256];    fc3: [256,2]  -> [16][256] zero-padded.
// ============================================================================
__global__ __launch_bounds__(256) void pack_weights_kernel(
    const float* __restrict__ Wl2, const float* __restrict__ Wr2,
    const float* __restrict__ fc1w, const float* __restrict__ fc2w,
    const float* __restrict__ fc3w,
    _Float16* __restrict__ Wl2T, _Float16* __restrict__ Wr2T,
    _Float16* __restrict__ fc1T, _Float16* __restrict__ fc2T,
    _Float16* __restrict__ fc3T) {
  int idx = blockIdx.x * 256 + threadIdx.x;
  if (idx < 16384) {                       // Wl2T[n][k] = Wl2[k][n]
    int n = idx >> 8, k = idx & 255;
    Wl2T[n * 256 + k] = (_Float16)Wl2[k * 64 + n];
    return;
  }
  idx -= 16384;
  if (idx < 16384) {                       // Wr2T
    int n = idx >> 8, k = idx & 255;
    Wr2T[n * 256 + k] = (_Float16)Wr2[k * 64 + n];
    return;
  }
  idx -= 16384;
  if (idx < 16384) {                       // fc1T[n][k] = fc1w[k][n]
    int n = idx >> 6, k = idx & 63;
    fc1T[n * 64 + k] = (_Float16)fc1w[k * 256 + n];
    return;
  }
  idx -= 16384;
  if (idx < 65536) {                       // fc2T
    int n = idx >> 8, k = idx & 255;
    fc2T[n * 256 + k] = (_Float16)fc2w[k * 256 + n];
    return;
  }
  idx -= 65536;
  if (idx < 4096) {                        // fc3T, rows 2..15 zero
    int n = idx >> 8, k = idx & 255;
    fc3T[n * 256 + k] = (n < 2) ? (_Float16)fc3w[k * 2 + n] : (_Float16)0.f;
  }
}

// ============================================================================
// Kernel 1: one workgroup per batch element (8 waves).
//   nodes -> GATv2 layer1 (VALU) -> layer2 linear via WMMA ->
//   head-1 attention + pool (collapsed) -> nfv[b][64]
// ============================================================================
__global__ __launch_bounds__(256) void gat_fused_kernel(
    const float* __restrict__ state24,
    const float* __restrict__ Wl1, const float* __restrict__ Wr1,
    const float* __restrict__ att1, const float* __restrict__ b1,
    const _Float16* __restrict__ Wl2T, const _Float16* __restrict__ Wr2T,
    const float* __restrict__ att2, const float* __restrict__ b2,
    float* __restrict__ nfv_out) {
  const int b = blockIdx.x;
  const int tid = threadIdx.x;

  // Phase-aliased LDS arena (2 x 21504 bytes):
  //   phase 1: regionA = gl1[21][256] f32, regionB = gr1[21][256] f32
  //   phase 2: regionB = h1h[32][264] f16 (layer1 out, padded for WMMA A)
  //   phase 3: regionA = gl2[32][68] f32 | gr2[32][68] f32 (WMMA D tiles)
  __shared__ __align__(16) char smem[43008];
  float*    gl1 = (float*)smem;                 // [21][256]
  float*    gr1 = (float*)(smem + 21504);       // [21][256]
  float*    gl2 = (float*)smem;                 // [32][68]
  float*    gr2 = (float*)(smem + 8704);        // [32][68]
  _Float16* h1h = (_Float16*)(smem + 21504);    // [32][264]

  __shared__ float xnode[NNODES * 7];
  __shared__ float e1[NNODES * NNODES * 4];     // scores/alpha; reused as e2
  __shared__ float att1s[256];
  __shared__ float att2s[64];
  __shared__ float scol[NNODES];

  // ---- build node features --------------------------------------------------
  if (tid < NNODES) {
    if (tid < 20) {
      float step = (PI_F + 0.03f) / 20.0f;
      float ang = -PI_F * 0.5f - 0.03f + (float)tid * step + PI_F / 20.0f;
      xnode[tid * 7 + 0] = state24[b * 24 + tid] * 0.1f;
      xnode[tid * 7 + 1] = sinf(ang);
      xnode[tid * 7 + 2] = cosf(ang);
      xnode[tid * 7 + 3] = 0.f; xnode[tid * 7 + 4] = 0.f;
      xnode[tid * 7 + 5] = 0.f; xnode[tid * 7 + 6] = 0.f;
    } else {  // robot node
      xnode[tid * 7 + 0] = 0.f; xnode[tid * 7 + 1] = 0.f; xnode[tid * 7 + 2] = 0.f;
#pragma unroll
      for (int q = 0; q < 4; ++q)
        xnode[tid * 7 + 3 + q] = state24[b * 24 + 20 + q];
    }
  }
  att1s[tid] = att1[tid];
  if (tid < 64) att2s[tid] = att2[tid];
  __syncthreads();

  // ---- layer1 source/target transforms (K=7, VALU) --------------------------
  {
    const int c = tid;
    for (int n = 0; n < NNODES; ++n) {
      float al = 0.f, ar = 0.f;
#pragma unroll
      for (int f = 0; f < 7; ++f) {
        float xv = xnode[n * 7 + f];
        al = fmaf(xv, Wl1[f * 256 + c], al);
        ar = fmaf(xv, Wr1[f * 256 + c], ar);
      }
      gl1[n * 256 + c] = al;
      gr1[n * 256 + c] = ar;
    }
  }
  __syncthreads();

  // ---- layer1 attention scores ----------------------------------------------
  for (int idx = tid; idx < NNODES * NNODES * 4; idx += 256) {
    int i = idx / 84, rem = idx % 84, j = rem / 4, h = rem & 3;
    const float* glv = &gl1[j * 256 + h * 64];
    const float* grv = &gr1[i * 256 + h * 64];
    const float* at  = &att1s[h * 64];
    float e = 0.f;
    for (int d = 0; d < 64; ++d) {
      float s = glv[d] + grv[d];
      s = s > 0.f ? s : 0.2f * s;
      e = fmaf(at[d], s, e);
    }
    e1[(i * NNODES + j) * 4 + h] = e;
  }
  __syncthreads();

  // ---- softmax over sources j per (i,h) -------------------------------------
  if (tid < 84) {
    int i = tid >> 2, h = tid & 3;
    float mx = -1e30f;
    for (int j = 0; j < NNODES; ++j) mx = fmaxf(mx, e1[(i * NNODES + j) * 4 + h]);
    float s = 0.f;
    for (int j = 0; j < NNODES; ++j) {
      float v = __expf(e1[(i * NNODES + j) * 4 + h] - mx);
      e1[(i * NNODES + j) * 4 + h] = v;
      s += v;
    }
    float inv = 1.f / s;
    for (int j = 0; j < NNODES; ++j) e1[(i * NNODES + j) * 4 + h] *= inv;
  }
  __syncthreads();

  // ---- aggregate + bias + ELU -> h1h (f16, rows padded to 32) ---------------
  {
    const int c = tid, h = c >> 6;
    const float bias = b1[c];
    for (int i = 0; i < NNODES; ++i) {
      float acc = 0.f;
      for (int j = 0; j < NNODES; ++j)
        acc = fmaf(e1[(i * NNODES + j) * 4 + h], gl1[j * 256 + c], acc);
      acc += bias;
      acc = acc > 0.f ? acc : (__expf(acc) - 1.f);
      h1h[i * 264 + c] = (_Float16)acc;
    }
    for (int i = NNODES; i < 32; ++i) h1h[i * 264 + c] = (_Float16)0.f;
  }
  __syncthreads();

  // ---- layer2 linear transforms via WMMA ------------------------------------
  // 8 waves: wave w -> {Wl2|Wr2} x n-tile(16 cols), both 16-row M-tiles, K=256.
  {
    const int lane  = tid & 31;
    const int w     = tid >> 5;
    const int which = w >> 2;            // 0: gl2 = h1@Wl2, 1: gr2 = h1@Wr2
    const int n0    = (w & 3) * 16;
    const _Float16* WT = which ? Wr2T : Wl2T;
    float* dst         = which ? gr2 : gl2;
    const int m = lane & 15, g = lane >> 4;

    v8f acc0 = {}, acc1 = {};
    for (int kc = 0; kc < 8; ++kc) {
      const int kb = kc * 32;
      v16h bf  = load_b_t16(WT, 256, kb, n0 + m, g);
      v16h af0 = load_a_lds(h1h, m,      264, kb, g);
      v16h af1 = load_a_lds(h1h, m + 16, 264, kb, g);
      acc0 = wmma_f16(af0, bf, acc0);
      acc1 = wmma_f16(af1, bf, acc1);
    }
#pragma unroll
    for (int r = 0; r < 8; ++r) {
      dst[(r + 8 * g) * 68      + n0 + m] = acc0[r];
      dst[(16 + r + 8 * g) * 68 + n0 + m] = acc1[r];
    }
  }
  __syncthreads();

  // ---- layer2 attention (1 head) --------------------------------------------
  for (int idx = tid; idx < NNODES * NNODES; idx += 256) {
    int i = idx / NNODES, j = idx % NNODES;
    float e = 0.f;
    for (int d = 0; d < 64; ++d) {
      float s = gl2[j * 68 + d] + gr2[i * 68 + d];
      s = s > 0.f ? s : 0.2f * s;
      e = fmaf(att2s[d], s, e);
    }
    e1[i * NNODES + j] = e;  // e1 reused as e2
  }
  __syncthreads();
  if (tid < NNODES) {
    const int i = tid;
    float mx = -1e30f;
    for (int j = 0; j < NNODES; ++j) mx = fmaxf(mx, e1[i * NNODES + j]);
    float s = 0.f;
    for (int j = 0; j < NNODES; ++j) {
      float v = __expf(e1[i * NNODES + j] - mx);
      e1[i * NNODES + j] = v;
      s += v;
    }
    float inv = 1.f / s;
    for (int j = 0; j < NNODES; ++j) e1[i * NNODES + j] *= inv;
  }
  __syncthreads();
  // pool collapse: mean_i sum_j alpha_ij gl2_j = (sum_j colsum_j gl2_j)/21
  if (tid < NNODES) {
    const int j = tid;
    float s = 0.f;
    for (int i = 0; i < NNODES; ++i) s += e1[i * NNODES + j];
    scol[j] = s;
  }
  __syncthreads();
  if (tid < 64) {
    const int d = tid;
    float acc = 0.f;
    for (int j = 0; j < NNODES; ++j) acc = fmaf(scol[j], gl2[j * 68 + d], acc);
    nfv_out[b * 64 + d] = acc * (1.0f / 21.0f) + b2[d];
  }
}

// ============================================================================
// Kernel 2: batched MLP, one wave per 16-row M-tile, all GEMMs via WMMA.
//   [B,64] -> relu fc1 [B,256] -> relu fc2 [B,256] -> tanh fc3 [B,2]
// ============================================================================
__global__ __launch_bounds__(32) void mlp_wmma_kernel(
    const float* __restrict__ nfv, int Btotal,
    const _Float16* __restrict__ fc1T, const float* __restrict__ fc1_b,
    const _Float16* __restrict__ fc2T, const float* __restrict__ fc2_b,
    const _Float16* __restrict__ fc3T, const float* __restrict__ fc3_b,
    float* __restrict__ out) {
  const int m0   = blockIdx.x * 16;
  const int lane = threadIdx.x;
  const int m = lane & 15, g = lane >> 4;

  __shared__ __align__(16) _Float16 a0h[16 * 72];    // input [16][64] (+pad)
  __shared__ __align__(16) _Float16 h1s[16 * 264];   // fc1 out [16][256] (+pad)
  __shared__ __align__(16) _Float16 h2s[16 * 264];   // fc2 out [16][256] (+pad)

  for (int idx = lane; idx < 16 * 64; idx += 32) {
    int r = idx >> 6, c = idx & 63;
    int row = m0 + r;
    a0h[r * 72 + c] = (_Float16)(row < Btotal ? nfv[row * 64 + c] : 0.f);
  }
  __syncthreads();

  // ---- fc1: [16,64] @ [64,256], relu ----------------------------------------
  for (int nt = 0; nt < 16; ++nt) {
    const int n0 = nt * 16;
    v8f acc = {};
#pragma unroll
    for (int kc = 0; kc < 2; ++kc) {
      const int kb = kc * 32;
      v16h af = load_a_lds(a0h, m, 72, kb, g);
      v16h bf = load_b_t16(fc1T, 64, kb, n0 + m, g);
      acc = wmma_f16(af, bf, acc);
    }
    const float bias = fc1_b[n0 + m];
#pragma unroll
    for (int r = 0; r < 8; ++r) {
      float v = acc[r] + bias;
      h1s[(r + 8 * g) * 264 + n0 + m] = (_Float16)(v > 0.f ? v : 0.f);
    }
  }
  __syncthreads();

  // ---- fc2: [16,256] @ [256,256], relu --------------------------------------
  for (int nt = 0; nt < 16; ++nt) {
    const int n0 = nt * 16;
    v8f acc = {};
    for (int kc = 0; kc < 8; ++kc) {
      const int kb = kc * 32;
      v16h af = load_a_lds(h1s, m, 264, kb, g);
      v16h bf = load_b_t16(fc2T, 256, kb, n0 + m, g);
      acc = wmma_f16(af, bf, acc);
    }
    const float bias = fc2_b[n0 + m];
#pragma unroll
    for (int r = 0; r < 8; ++r) {
      float v = acc[r] + bias;
      h2s[(r + 8 * g) * 264 + n0 + m] = (_Float16)(v > 0.f ? v : 0.f);
    }
  }
  __syncthreads();

  // ---- fc3: [16,256] @ [256,2] (N zero-padded to 16 in fc3T), tanh ----------
  {
    v8f acc = {};
    for (int kc = 0; kc < 8; ++kc) {
      const int kb = kc * 32;
      v16h af = load_a_lds(h2s, m, 264, kb, g);
      v16h bf = load_b_t16(fc3T, 256, kb, m, g);
      acc = wmma_f16(af, bf, acc);
    }
    if (m < 2) {
      const float bias = fc3_b[m];
#pragma unroll
      for (int r = 0; r < 8; ++r) {
        int row = m0 + r + 8 * g;
        if (row < Btotal) out[row * 2 + m] = tanhf(acc[r] + bias);
      }
    }
  }
}

// ============================================================================
extern "C" void kernel_launch(void* const* d_in, const int* in_sizes, int n_in,
                              void* d_out, int out_size, void* d_ws, size_t ws_size,
                              hipStream_t stream) {
  const float* state24 = (const float*)d_in[0];
  const float* Wl1  = (const float*)d_in[1];
  const float* Wr1  = (const float*)d_in[2];
  const float* att1 = (const float*)d_in[3];
  const float* b1   = (const float*)d_in[4];
  const float* Wl2  = (const float*)d_in[5];
  const float* Wr2  = (const float*)d_in[6];
  const float* att2 = (const float*)d_in[7];
  const float* b2   = (const float*)d_in[8];
  const float* fc1_w = (const float*)d_in[9];
  const float* fc1_b = (const float*)d_in[10];
  const float* fc2_w = (const float*)d_in[11];
  const float* fc2_b = (const float*)d_in[12];
  const float* fc3_w = (const float*)d_in[13];
  const float* fc3_b = (const float*)d_in[14];

  const int B = in_sizes[0] / 24;

  // workspace layout: nfv f32 [B][64], then packed f16 transposed weights
  float* nfv = (float*)d_ws;
  size_t off = ((size_t)B * 64 * sizeof(float) + 255) & ~(size_t)255;
  _Float16* Wl2T = (_Float16*)((char*)d_ws + off);   // [64][256]
  _Float16* Wr2T = Wl2T + 64 * 256;                  // [64][256]
  _Float16* fc1T = Wr2T + 64 * 256;                  // [256][64]
  _Float16* fc2T = fc1T + 256 * 64;                  // [256][256]
  _Float16* fc3T = fc2T + 256 * 256;                 // [16][256] (padded)

  // 16384*3 + 65536 + 4096 = 118784 elements to pack
  pack_weights_kernel<<<(118784 + 255) / 256, 256, 0, stream>>>(
      Wl2, Wr2, fc1_w, fc2_w, fc3_w, Wl2T, Wr2T, fc1T, fc2T, fc3T);

  gat_fused_kernel<<<B, 256, 0, stream>>>(state24, Wl1, Wr1, att1, b1,
                                          Wl2T, Wr2T, att2, b2, nfv);

  mlp_wmma_kernel<<<(B + 15) / 16, 32, 0, stream>>>(nfv, B,
                                                    fc1T, fc1_b, fc2T, fc2_b,
                                                    fc3T, fc3_b, (float*)d_out);
}
